// AttentionHead_13572096656163
// MI455X (gfx1250) — matrix operand order repaired and checked by reading, not compiled, not run on hardware
//
#include <hip/hip_runtime.h>
#include <hip/hip_fp16.h>

typedef __attribute__((ext_vector_type(16))) _Float16 v16h;
typedef __attribute__((ext_vector_type(8)))  _Float16 v8h;
typedef __attribute__((ext_vector_type(8)))  float    v8f;

#define N_EMBD 768
#define HEAD   64
#define TSEQ   4096
#define BATCH  4
#define BT     (BATCH * TSEQ)

static __device__ __forceinline__ float fast_exp2(float x) {
  return __builtin_amdgcn_exp2f(x);   // v_exp_f32
}

// ---------------------------------------------------------------------------
// Kernel 0: W [768][64] f32 (row-major)  ->  Wt [64][768] f16 (transposed)
// so that WMMA B-fragments of W are contiguous 32B runs.
// ---------------------------------------------------------------------------
__global__ void k_convw(const float* __restrict__ W, _Float16* __restrict__ Wt) {
  int i = blockIdx.x * blockDim.x + threadIdx.x;   // over 768*64
  if (i < N_EMBD * HEAD) {
    int c = i >> 6;       // embed dim
    int n = i & 63;       // head dim
    Wt[n * N_EMBD + c] = (_Float16)W[i];
  }
}

// ---------------------------------------------------------------------------
// Kernel 1: QKV projection with V_WMMA_F32_16X16X32_F16.
// One wave32 per 16 rows of x. Outputs:
//   q, k : f16 row-major [BT][64]
//   vT   : f16 [B][64][T]   (transposed so PV B-fragments are contiguous)
// ---------------------------------------------------------------------------
__global__ void __launch_bounds__(32)
k_qkv(const float* __restrict__ x,
      const _Float16* __restrict__ Wtq,
      const _Float16* __restrict__ Wtk,
      const _Float16* __restrict__ Wtv,
      _Float16* __restrict__ q,
      _Float16* __restrict__ k,
      _Float16* __restrict__ vT) {
  const int lane = threadIdx.x & 31;
  const int r16  = lane & 15;
  const int half = lane >> 4;

  const long arow = (long)blockIdx.x * 16 + r16;
  const float* xrow = x + arow * N_EMBD;

  v8f accq[4], acck[4], accv[4];
#pragma unroll
  for (int nt = 0; nt < 4; ++nt)
#pragma unroll
    for (int j = 0; j < 8; ++j) { accq[nt][j] = 0.f; acck[nt][j] = 0.f; accv[nt][j] = 0.f; }

  for (int kk = 0; kk < N_EMBD; kk += 32) {
    // A fragment (16x32 f16): j=0..15 -> K=kk+half*16+j
    const float* p0 = xrow + kk + half * 16;
    const float* p1 = xrow + kk + half * 16 + 8;
    float4 r0 = *(const float4*)(p0);
    float4 r1 = *(const float4*)(p0 + 4);
    float4 r2 = *(const float4*)(p1);
    float4 r3 = *(const float4*)(p1 + 4);
    v16h a;
    a[0]  = (_Float16)r0.x; a[1]  = (_Float16)r0.y; a[2]  = (_Float16)r0.z; a[3]  = (_Float16)r0.w;
    a[4]  = (_Float16)r1.x; a[5]  = (_Float16)r1.y; a[6]  = (_Float16)r1.z; a[7]  = (_Float16)r1.w;
    a[8]  = (_Float16)r2.x; a[9]  = (_Float16)r2.y; a[10] = (_Float16)r2.z; a[11] = (_Float16)r2.w;
    a[12] = (_Float16)r3.x; a[13] = (_Float16)r3.y; a[14] = (_Float16)r3.z; a[15] = (_Float16)r3.w;

#pragma unroll
    for (int nt = 0; nt < 4; ++nt) {
      const int n = nt * 16 + r16;                            // B column
      const long boff = (long)n * N_EMBD + kk + 16 * half;    // K = j + 16*half
      v16h bq = *(const v16h*)(Wtq + boff);
      v16h bk = *(const v16h*)(Wtk + boff);
      v16h bv = *(const v16h*)(Wtv + boff);
      accq[nt] = __builtin_amdgcn_wmma_f32_16x16x32_f16(false, a, false, bq, (short)0, accq[nt], false, false);
      acck[nt] = __builtin_amdgcn_wmma_f32_16x16x32_f16(false, a, false, bk, (short)0, acck[nt], false, false);
      accv[nt] = __builtin_amdgcn_wmma_f32_16x16x32_f16(false, a, false, bv, (short)0, accv[nt], false, false);
    }
  }

#pragma unroll
  for (int nt = 0; nt < 4; ++nt) {
#pragma unroll
    for (int v = 0; v < 8; ++v) {
      const int  M = v + 8 * half;
      const long t = (long)blockIdx.x * 16 + M;
      const int  h = nt * 16 + r16;
      q[t * HEAD + h] = (_Float16)accq[nt][v];
      k[t * HEAD + h] = (_Float16)acck[nt][v];
      const int b  = (int)(t >> 12);
      const int tl = (int)(t & (TSEQ - 1));
      vT[((long)(b * HEAD + h)) * TSEQ + tl] = (_Float16)accv[nt][v];
    }
  }
}

// ---------------------------------------------------------------------------
// Kernel 2: causal flash-attention, 4 waves per block (128 threads).
// Each wave owns an interleaved quarter of the key stream (s0 = 32*wave,
// stride 128), runs its own fp32 online softmax (exp2 domain), then the four
// partial (O, m, l) sets are merged through LDS.
// Per 32-key step per wave: 4 WMMAs for S = Q K^T, 4 WMMAs for O += P V.
// ---------------------------------------------------------------------------
__global__ void __launch_bounds__(128)
k_attn(const _Float16* __restrict__ q,
       const _Float16* __restrict__ k,
       const _Float16* __restrict__ vT,
       float* __restrict__ out) {
  __shared__ float    sO[4][16][64];     // per-wave partial O (16 KB)
  __shared__ float    sM[4][16];         // per-wave row maxima (log2 domain)
  __shared__ float    sL[4][16];         // per-wave row sums
  __shared__ _Float16 pl[4][16][40];     // per-wave P staging, padded rows

  const int tid  = threadIdx.x;
  const int wave = tid >> 5;
  const int lane = tid & 31;
  const int r16  = lane & 15;
  const int half = lane >> 4;

  const int qrow0 = blockIdx.x * 16;     // global query row base (over B*T)
  const int b     = qrow0 >> 12;
  const int tq0   = qrow0 & (TSEQ - 1);

  const float NEG_INF = -__builtin_inff();
  const float M_FLOOR = -1.0e30f;                      // finite: avoids NaN in empty waves
  const float SCALE   = 0.125f * 1.44269504088896f;    // 1/sqrt(64) * log2(e)

  // ---- Q A-fragments (K-dim 0..63 in two fragments), register-resident
  v16h qa[2];
  {
    const _Float16* qrow = q + (long)(qrow0 + r16) * HEAD;
#pragma unroll
    for (int kf = 0; kf < 2; ++kf) {
      v8h lo = *(const v8h*)(qrow + kf * 32 + half * 16);
      v8h hi = *(const v8h*)(qrow + kf * 32 + half * 16 + 8);
      v16h a;
#pragma unroll
      for (int j = 0; j < 8; ++j) { a[j] = lo[j]; a[8 + j] = hi[j]; }
      qa[kf] = a;
    }
  }

  v8f oacc[4];
#pragma unroll
  for (int nt = 0; nt < 4; ++nt)
#pragma unroll
    for (int j = 0; j < 8; ++j) oacc[nt][j] = 0.f;

  float m[8], l[8];
#pragma unroll
  for (int v = 0; v < 8; ++v) { m[v] = M_FLOOR; l[v] = 0.f; }

  const _Float16* kbase = k  + (long)b * TSEQ * HEAD;
  const _Float16* vbase = vT + (long)b * HEAD * TSEQ;

  const int send = tq0 + 16;             // keys needed: [0, tq0+15]
  for (int s0 = 32 * wave; s0 < send; s0 += 128) {
    // ---- scores: two 16x16 tiles covering keys s0..s0+31
    v8f S[2];
#pragma unroll
    for (int c = 0; c < 2; ++c) {
      const int s = s0 + c * 16 + r16;   // B column = key row
      const _Float16* krow = kbase + (long)s * HEAD;
      if (s + 128 < TSEQ) __builtin_prefetch(krow + 128 * HEAD, 0, 1);
      v16h b0 = *(const v16h*)(krow + 16 * half);
      v16h b1 = *(const v16h*)(krow + 32 + 16 * half);
      v8f acc;
#pragma unroll
      for (int j = 0; j < 8; ++j) acc[j] = 0.f;
      acc = __builtin_amdgcn_wmma_f32_16x16x32_f16(false, qa[0], false, b0, (short)0, acc, false, false);
      acc = __builtin_amdgcn_wmma_f32_16x16x32_f16(false, qa[1], false, b1, (short)0, acc, false, false);
      S[c] = acc;
    }

    // ---- scale (log2 domain), causal mask, row max
    float mnew[8];
#pragma unroll
    for (int v = 0; v < 8; ++v) {
      const int t  = tq0 + v + 8 * half;
      float s0v = S[0][v] * SCALE;
      float s1v = S[1][v] * SCALE;
      if (s0 + r16 > t)      s0v = NEG_INF;
      if (s0 + 16 + r16 > t) s1v = NEG_INF;
      S[0][v] = s0v; S[1][v] = s1v;
      float rm = fmaxf(s0v, s1v);
      rm = fmaxf(rm, __shfl_xor(rm, 1, 32));
      rm = fmaxf(rm, __shfl_xor(rm, 2, 32));
      rm = fmaxf(rm, __shfl_xor(rm, 4, 32));
      rm = fmaxf(rm, __shfl_xor(rm, 8, 32));
      mnew[v] = fmaxf(m[v], rm);         // >= M_FLOOR always (finite)
    }

    // ---- p = 2^(s - m_new), row sums, online rescale
    float fac[8];
#pragma unroll
    for (int v = 0; v < 8; ++v) {
      float p0 = fast_exp2(S[0][v] - mnew[v]);
      float p1 = fast_exp2(S[1][v] - mnew[v]);
      S[0][v] = p0; S[1][v] = p1;
      float rs = p0 + p1;
      rs += __shfl_xor(rs, 1, 32);
      rs += __shfl_xor(rs, 2, 32);
      rs += __shfl_xor(rs, 4, 32);
      rs += __shfl_xor(rs, 8, 32);
      fac[v] = fast_exp2(m[v] - mnew[v]);
      l[v]   = l[v] * fac[v] + rs;
      m[v]   = mnew[v];
    }
#pragma unroll
    for (int nt = 0; nt < 4; ++nt)
#pragma unroll
      for (int v = 0; v < 8; ++v) oacc[nt][v] *= fac[v];

    // ---- stage P (C-layout) through wave-private LDS, read back as A-frag.
    // DS ops from one wave complete in order; a raw s_wait_dscnt plus a
    // compiler memory barrier is sufficient (no cross-wave barrier legal here:
    // waves have different trip counts).
    asm volatile("" ::: "memory");
#pragma unroll
    for (int c = 0; c < 2; ++c)
#pragma unroll
      for (int v = 0; v < 8; ++v)
        pl[wave][v + 8 * half][c * 16 + r16] = (_Float16)S[c][v];
    asm volatile("s_wait_dscnt 0" ::: "memory");

    v16h pa;
    {
      v8h lo = *(const v8h*)&pl[wave][r16][half * 16];
      v8h hi = *(const v8h*)&pl[wave][r16][half * 16 + 8];
#pragma unroll
      for (int j = 0; j < 8; ++j) { pa[j] = lo[j]; pa[8 + j] = hi[j]; }
    }
    asm volatile("" ::: "memory");

    // ---- O += P @ V   (B from vT: contiguous 16 halves per lane)
#pragma unroll
    for (int nt = 0; nt < 4; ++nt) {
      const _Float16* vrow = vbase + (long)(nt * 16 + r16) * TSEQ + s0 + 16 * half;
      if (s0 + 128 < TSEQ) __builtin_prefetch(vrow + 128, 0, 1);
      v16h vb = *(const v16h*)vrow;
      oacc[nt] = __builtin_amdgcn_wmma_f32_16x16x32_f16(false, pa, false, vb, (short)0, oacc[nt], false, false);
    }
  }

  // ---- publish per-wave partials
#pragma unroll
  for (int nt = 0; nt < 4; ++nt)
#pragma unroll
    for (int v = 0; v < 8; ++v)
      sO[wave][v + 8 * half][nt * 16 + r16] = oacc[nt][v];
  if (r16 == 0) {
#pragma unroll
    for (int v = 0; v < 8; ++v) {
      sM[wave][v + 8 * half] = m[v];
      sL[wave][v + 8 * half] = l[v];
    }
  }
  __syncthreads();   // uniform: executed once by all 128 threads

  // ---- merge 4 partial softmaxes, write output (128 threads x 8 elements)
  for (int e = tid; e < 16 * HEAD; e += 128) {
    const int r = e >> 6;        // query row within tile
    const int h = e & 63;        // head dim
    const float m0 = sM[0][r], m1 = sM[1][r], m2 = sM[2][r], m3 = sM[3][r];
    const float Mf = fmaxf(fmaxf(m0, m1), fmaxf(m2, m3));
    const float w0 = fast_exp2(m0 - Mf);
    const float w1 = fast_exp2(m1 - Mf);
    const float w2 = fast_exp2(m2 - Mf);
    const float w3 = fast_exp2(m3 - Mf);
    const float L = sL[0][r] * w0 + sL[1][r] * w1 + sL[2][r] * w2 + sL[3][r] * w3;
    const float O = sO[0][r][h] * w0 + sO[1][r][h] * w1 + sO[2][r][h] * w2 + sO[3][r][h] * w3;
    out[((long)(b * TSEQ + tq0 + r)) * HEAD + h] = O / L;
  }
}

// ---------------------------------------------------------------------------
// Host launcher. Workspace layout (f16 elements, all offsets 32B-aligned):
//   q / k : BT*64 halves (2 MB each),  vT : B*64*T halves (2 MB)
//   WtQ/WtK/WtV : 64*768 halves each (96 KB each).  Total ~6.3 MB of d_ws.
// ---------------------------------------------------------------------------
extern "C" void kernel_launch(void* const* d_in, const int* in_sizes, int n_in,
                              void* d_out, int out_size, void* d_ws, size_t ws_size,
                              hipStream_t stream) {
  (void)in_sizes; (void)n_in; (void)out_size; (void)ws_size;

  const float* x  = (const float*)d_in[0];
  const float* Wk = (const float*)d_in[1];
  const float* Wq = (const float*)d_in[2];
  const float* Wv = (const float*)d_in[3];
  float* out = (float*)d_out;

  _Float16* ws  = (_Float16*)d_ws;
  _Float16* qf  = ws;
  _Float16* kf  = ws + (size_t)BT * HEAD;
  _Float16* vt  = ws + (size_t)2 * BT * HEAD;
  _Float16* wtq = ws + (size_t)3 * BT * HEAD;
  _Float16* wtk = wtq + (size_t)N_EMBD * HEAD;
  _Float16* wtv = wtk + (size_t)N_EMBD * HEAD;

  k_convw<<<(N_EMBD * HEAD + 255) / 256, 256, 0, stream>>>(Wq, wtq);
  k_convw<<<(N_EMBD * HEAD + 255) / 256, 256, 0, stream>>>(Wk, wtk);
  k_convw<<<(N_EMBD * HEAD + 255) / 256, 256, 0, stream>>>(Wv, wtv);

  // QKV projection: 1024 blocks x 1 wave, 16 rows each
  k_qkv<<<BT / 16, 32, 0, stream>>>(x, wtq, wtk, wtv, qf, kf, vt);

  // Flash attention: 1024 blocks x 4 waves (key-split), 16 queries each
  k_attn<<<BT / 16, 128, 0, stream>>>(qf, kf, vt, out);
}
